// CausalSelfAttention_5798205849912
// MI455X (gfx1250) — compile-verified, matrix operand order checked by reading, not run on hardware
//
#include <hip/hip_runtime.h>
#include <hip/hip_bf16.h>

// ---------------------------------------------------------------------------
// Problem constants (from reference): B=8, T=1024, C=768, NH=12, HD=64
// ---------------------------------------------------------------------------
constexpr int Bsz = 8;
constexpr int T   = 1024;
constexpr int C   = 768;
constexpr int NH  = 12;
constexpr int HD  = 64;
constexpr int C3  = 3 * C;          // 2304
constexpr int M   = Bsz * T;        // 8192 rows for the projections
constexpr int BH  = Bsz * NH;       // 96 (batch*heads)

typedef __attribute__((ext_vector_type(16))) __bf16       v16bf;
typedef __attribute__((ext_vector_type(8)))  float        v8f;
typedef __attribute__((ext_vector_type(4)))  unsigned int v4u;

#define WMMA_BF16(a, b, c) \
    __builtin_amdgcn_wmma_f32_16x16x32_bf16(false, (a), false, (b), (short)0, (c), false, false)

// float -> bf16 (round to nearest even)
__device__ __forceinline__ unsigned short f2bf(float f) {
    unsigned int u = __float_as_uint(f);
    return (unsigned short)((u + 0x7FFFu + ((u >> 16) & 1u)) >> 16);
}

// Load one 16-lane x 16-bf16 fragment: 8 contiguous bf16 at p, 8 more at p+16.
// Matches the CDNA5 16-bit A/B fragment layout (K 0..7 in v0..3, K 16..23 in v4..7).
__device__ __forceinline__ v16bf load_frag(const unsigned short* p) {
    union { v4u u[2]; v16bf b; } t;
    t.u[0] = *(const v4u*)(p);
    t.u[1] = *(const v4u*)(p + 16);
    return t.b;
}

// A-fragment (16x32, row-major source, stride ld) at (r0, k0).
// Lane l (0..15): row r0+l, K = k0+{0..7, 16..23}; lane l+16: same row set with K+8/+24.
// B-fragment is the mirror with "rows" = output columns, so the same loader works on
// an N-major (transposed) B.
__device__ __forceinline__ v16bf load_tile16(const unsigned short* base, int ld, int r0, int k0) {
    int lane = threadIdx.x & 31;
    int r    = lane & 15;
    int h    = lane >> 4;
    return load_frag(base + (size_t)(r0 + r) * ld + k0 + h * 8);
}

// ---------------------------------------------------------------------------
// Elementwise converts
// ---------------------------------------------------------------------------
__global__ __launch_bounds__(256) void k_cvt_bf16(const float* __restrict__ src,
                                                  unsigned short* __restrict__ dst, int n) {
    int i = blockIdx.x * 256 + threadIdx.x;
    if (i < n) dst[i] = f2bf(src[i]);
}

// dst[n*K + k] = bf16(src[k*N + n])  (weights -> N-major bf16)
__global__ __launch_bounds__(256) void k_cvt_tr_bf16(const float* __restrict__ src,
                                                     unsigned short* __restrict__ dst,
                                                     int K, int N) {
    int i = blockIdx.x * 256 + threadIdx.x;
    if (i >= K * N) return;
    int k = i % K;
    int n = i / K;
    dst[i] = f2bf(src[(size_t)k * N + n]);
}

// ---------------------------------------------------------------------------
// 64x64-per-wave WMMA GEMM core.  A: [M,K] bf16 row-major (lda), Bt: [N,K] bf16
// row-major (ldb).  Produces 4x4 C fragments of 16x16 f32.
// ---------------------------------------------------------------------------
__device__ __forceinline__ void gemm_core_64x64(const unsigned short* __restrict__ A,
                                                const unsigned short* __restrict__ Bt,
                                                int lda, int ldb, int K,
                                                int row0, int col0, v8f acc[4][4]) {
    int lane = threadIdx.x & 31;
    for (int kt = 0; kt < K; kt += 32) {
        v16bf a[4], b[4];
#pragma unroll
        for (int i = 0; i < 4; ++i) a[i] = load_tile16(A, lda, row0 + 16 * i, kt);
#pragma unroll
        for (int j = 0; j < 4; ++j) b[j] = load_tile16(Bt, ldb, col0 + 16 * j, kt);
        // prefetch next k-slab (global_prefetch_b8; speculative, never faults)
        __builtin_prefetch(A + (size_t)(row0 + (lane & 15)) * lda + kt + 64, 0, 1);
        __builtin_prefetch(Bt + (size_t)(col0 + (lane & 15)) * ldb + kt + 64, 0, 1);
#pragma unroll
        for (int i = 0; i < 4; ++i)
#pragma unroll
            for (int j = 0; j < 4; ++j)
                acc[i][j] = WMMA_BF16(a[i], b[j], acc[i][j]);
    }
}

// ---------------------------------------------------------------------------
// QKV projection: qkv = x @ w_attn + b_attn, scattered into head-major layouts:
//   Q, K : [BH, T, HD] bf16      V : [BH, HD, T] bf16 (transposed for PV B-frags)
// Block = 128 threads = 4 waves in a 2x2 arrangement -> 128x128 tile per block.
// Each 64-wide column tile lies inside exactly one of {q,k,v} and one head.
// ---------------------------------------------------------------------------
__global__ __launch_bounds__(128) void k_gemm_qkv(const unsigned short* __restrict__ A,
                                                  const unsigned short* __restrict__ Bt,
                                                  const float* __restrict__ bias,
                                                  unsigned short* __restrict__ qb,
                                                  unsigned short* __restrict__ kb,
                                                  unsigned short* __restrict__ vb) {
    int wave = threadIdx.x >> 5;
    int lane = threadIdx.x & 31;
    int row0 = blockIdx.x * 128 + (wave >> 1) * 64;
    int col0 = blockIdx.y * 128 + (wave & 1) * 64;

    v8f acc[4][4] = {};
    gemm_core_64x64(A, Bt, C, C, C, row0, col0, acc);

    int half  = lane >> 4;
    int c     = lane & 15;
    int which = col0 / C;              // 0=q 1=k 2=v (constant over the 64-wide tile)
    int head  = (col0 % C) / HD;       // constant over the tile
    int dbase = col0 % HD;             // 0 (col0 is a multiple of 64)

#pragma unroll
    for (int i = 0; i < 4; ++i) {
#pragma unroll
        for (int v = 0; v < 8; ++v) {
            int m  = row0 + 16 * i + v + 8 * half;   // global row
            int bb = m >> 10;                        // m / T
            int tt = m & 1023;                       // m % T
            int bh = bb * NH + head;
#pragma unroll
            for (int j = 0; j < 4; ++j) {
                int d   = dbase + 16 * j + c;        // 0..63 within head
                int n   = col0 + 16 * j + c;         // global output column
                float y = acc[i][j][v] + bias[n];
                unsigned short bv = f2bf(y);
                if (which == 0)      qb[((size_t)bh * T + tt) * HD + d] = bv;
                else if (which == 1) kb[((size_t)bh * T + tt) * HD + d] = bv;
                else                 vb[((size_t)bh * HD + d) * T + tt] = bv;
            }
        }
    }
}

// ---------------------------------------------------------------------------
// Causal flash attention.  One wave per 16-query tile; 4 waves per block.
// S = Q K^T (2 C-frags of 16x16 over 32 keys), online softmax in registers,
// P restaged through LDS (C-frag f32 -> A-frag bf16), O += P V via 4 WMMAs.
// Output written bf16 to y [B,T,C] for the projection GEMM.
// ---------------------------------------------------------------------------
__global__ __launch_bounds__(128) void k_attn(const unsigned short* __restrict__ q,
                                              const unsigned short* __restrict__ k,
                                              const unsigned short* __restrict__ vT,
                                              unsigned short* __restrict__ ybf) {
    __shared__ unsigned short ldsP[4][16 * 32];   // 1KB per wave

    int wave  = threadIdx.x >> 5;
    int lane  = threadIdx.x & 31;
    int half  = lane >> 4;
    int c     = lane & 15;

    int tile  = blockIdx.x * 4 + wave;            // 0 .. BH*(T/16)-1
    int qt    = tile & 63;                        // query tile within (b,h)
    int bh    = tile >> 6;
    int head  = bh % NH;
    int bb    = bh / NH;
    int qbase = qt * 16;

    const unsigned short* qh = q  + (size_t)bh * T * HD;
    const unsigned short* kh = k  + (size_t)bh * T * HD;
    const unsigned short* vh = vT + (size_t)bh * HD * T;
    unsigned short* pP = ldsP[wave];

    v16bf aq0 = load_tile16(qh, HD, qbase, 0);
    v16bf aq1 = load_tile16(qh, HD, qbase, 32);

    v8f o0 = {}, o1 = {}, o2 = {}, o3 = {};
    float mi[8], li[8];
#pragma unroll
    for (int v = 0; v < 8; ++v) { mi[v] = -3.0e38f; li[v] = 0.0f; }

    const float scale = 0.125f;                   // 1/sqrt(64)

    for (int kt = 0; kt <= qbase + 15; kt += 32) {
        // ---- S = Q K^T over 32 keys: two 16x16 C fragments -----------------
        v8f s0 = {}, s1 = {};
        v16bf bk;
        bk = load_tile16(kh, HD, kt,      0);  s0 = WMMA_BF16(aq0, bk, s0);
        bk = load_tile16(kh, HD, kt,     32);  s0 = WMMA_BF16(aq1, bk, s0);
        bk = load_tile16(kh, HD, kt + 16, 0);  s1 = WMMA_BF16(aq0, bk, s1);
        bk = load_tile16(kh, HD, kt + 16,32);  s1 = WMMA_BF16(aq1, bk, s1);

        // ---- online softmax (row reductions across the 16-lane col groups) -
        float alpha[8];
#pragma unroll
        for (int v = 0; v < 8; ++v) {
            int row = qbase + v + 8 * half;
            float x0 = s0[v] * scale;
            float x1 = s1[v] * scale;
            if (kt + c      > row) x0 = -3.0e38f;   // causal mask
            if (kt + 16 + c > row) x1 = -3.0e38f;
            float tmax = fmaxf(x0, x1);
#pragma unroll
            for (int msk = 1; msk <= 8; msk <<= 1)
                tmax = fmaxf(tmax, __shfl_xor(tmax, msk, 32));
            float mnew = fmaxf(mi[v], tmax);
            float a    = __expf(mi[v] - mnew);
            float p0   = __expf(x0 - mnew);
            float p1   = __expf(x1 - mnew);
            float rs   = p0 + p1;
#pragma unroll
            for (int msk = 1; msk <= 8; msk <<= 1)
                rs += __shfl_xor(rs, msk, 32);
            li[v]    = li[v] * a + rs;
            mi[v]    = mnew;
            alpha[v] = a;
            // stage P into LDS (16 rows x 32 keys, bf16)
            pP[(v + 8 * half) * 32 + c]      = f2bf(p0);
            pP[(v + 8 * half) * 32 + 16 + c] = f2bf(p1);
        }

        // rescale running output
#pragma unroll
        for (int v = 0; v < 8; ++v) {
            o0[v] *= alpha[v]; o1[v] *= alpha[v];
            o2[v] *= alpha[v]; o3[v] *= alpha[v];
        }

        // ---- O += P V : A-frag from LDS, B-frags from transposed V ---------
        v16bf aP = load_tile16(pP, 32, 0, 0);     // ds_load_b128 x2 per lane
        v16bf bv;
        bv = load_tile16(vh, T,  0, kt);  o0 = WMMA_BF16(aP, bv, o0);
        bv = load_tile16(vh, T, 16, kt);  o1 = WMMA_BF16(aP, bv, o1);
        bv = load_tile16(vh, T, 32, kt);  o2 = WMMA_BF16(aP, bv, o2);
        bv = load_tile16(vh, T, 48, kt);  o3 = WMMA_BF16(aP, bv, o3);
    }

    // ---- finalize: O /= l, write bf16 into y [B,T,C] -----------------------
#pragma unroll
    for (int v = 0; v < 8; ++v) {
        float inv = 1.0f / li[v];
        int tt = qbase + v + 8 * half;
        size_t ro = ((size_t)bb * T + tt) * C + head * HD;
        ybf[ro +  0 + c] = f2bf(o0[v] * inv);
        ybf[ro + 16 + c] = f2bf(o1[v] * inv);
        ybf[ro + 32 + c] = f2bf(o2[v] * inv);
        ybf[ro + 48 + c] = f2bf(o3[v] * inv);
    }
}

// ---------------------------------------------------------------------------
// Output projection: out = y @ w_proj + b_proj  (fp32 result)
// ---------------------------------------------------------------------------
__global__ __launch_bounds__(128) void k_gemm_proj(const unsigned short* __restrict__ A,
                                                   const unsigned short* __restrict__ Bt,
                                                   const float* __restrict__ bias,
                                                   float* __restrict__ out) {
    int wave = threadIdx.x >> 5;
    int lane = threadIdx.x & 31;
    int row0 = blockIdx.x * 128 + (wave >> 1) * 64;
    int col0 = blockIdx.y * 128 + (wave & 1) * 64;

    v8f acc[4][4] = {};
    gemm_core_64x64(A, Bt, C, C, C, row0, col0, acc);

    int half = lane >> 4;
    int c    = lane & 15;
#pragma unroll
    for (int i = 0; i < 4; ++i) {
#pragma unroll
        for (int v = 0; v < 8; ++v) {
            int m = row0 + 16 * i + v + 8 * half;
#pragma unroll
            for (int j = 0; j < 4; ++j) {
                int n = col0 + 16 * j + c;
                out[(size_t)m * C + n] = acc[i][j][v] + bias[n];
            }
        }
    }
}

// ---------------------------------------------------------------------------
// Host-side launcher
// ---------------------------------------------------------------------------
extern "C" void kernel_launch(void* const* d_in, const int* in_sizes, int n_in,
                              void* d_out, int out_size, void* d_ws, size_t ws_size,
                              hipStream_t stream) {
    const float* x      = (const float*)d_in[0];
    const float* w_attn = (const float*)d_in[1];
    const float* b_attn = (const float*)d_in[2];
    const float* w_proj = (const float*)d_in[3];
    const float* b_proj = (const float*)d_in[4];
    float*       out    = (float*)d_out;

    char*  ws  = (char*)d_ws;
    size_t off = 0;
    auto alloc = [&](size_t bytes) -> void* {
        void* p = ws + off;
        off = (off + bytes + 255) & ~(size_t)255;
        return p;
    };

    unsigned short* xbf    = (unsigned short*)alloc((size_t)M * C * 2);
    unsigned short* wattnT = (unsigned short*)alloc((size_t)C3 * C * 2);
    unsigned short* wprojT = (unsigned short*)alloc((size_t)C * C * 2);
    unsigned short* qb     = (unsigned short*)alloc((size_t)BH * T * HD * 2);
    unsigned short* kb     = (unsigned short*)alloc((size_t)BH * T * HD * 2);
    unsigned short* vb     = (unsigned short*)alloc((size_t)BH * HD * T * 2);
    unsigned short* ybf    = (unsigned short*)alloc((size_t)M * C * 2);
    (void)ws_size; (void)in_sizes; (void)n_in; (void)out_size;

    // 1) input / weight conversion to bf16 (weights N-major for B-fragments)
    k_cvt_bf16<<<dim3((M * C + 255) / 256), 256, 0, stream>>>(x, xbf, M * C);
    k_cvt_tr_bf16<<<dim3((C3 * C + 255) / 256), 256, 0, stream>>>(w_attn, wattnT, C, C3);
    k_cvt_tr_bf16<<<dim3((C * C + 255) / 256), 256, 0, stream>>>(w_proj, wprojT, C, C);

    // 2) fused QKV projection -> head-major Q/K + transposed V
    k_gemm_qkv<<<dim3(M / 128, C3 / 128), 128, 0, stream>>>(xbf, wattnT, b_attn, qb, kb, vb);

    // 3) causal flash attention -> y (bf16, [B,T,C])
    k_attn<<<dim3(BH * (T / 16) / 4), 128, 0, stream>>>(qb, kb, vb, ybf);

    // 4) output projection -> fp32 d_out
    k_gemm_proj<<<dim3(M / 128, C / 128), 128, 0, stream>>>(ybf, wprojT, b_proj, out);
}
